// SimpleEdgeInferModule_58566174048691
// MI455X (gfx1250) — matrix operand order repaired and checked by reading, not compiled, not run on hardware
//
#include <hip/hip_runtime.h>

typedef _Float16 v16h __attribute__((ext_vector_type(16)));
typedef _Float16 h8v  __attribute__((ext_vector_type(8)));
typedef float    v8f  __attribute__((ext_vector_type(8)));

static constexpr int Bc  = 256;
static constexpr int Nc  = 64;
static constexpr int Ec  = 4032;            // N*(N-1)
static constexpr int MT  = 128;             // rows per block tile
static constexpr size_t NODE_OUT_ELEMS = (size_t)Bc * Nc * 64;       // 1,048,576
static constexpr size_t EDGE_OUT_ELEMS = (size_t)Bc * Ec * 64;       // 66,060,288

// ---- workspace layout (bytes) ----
static constexpr size_t OFF_PKW   = 0;          // packed f16 weights (221184 B)
static constexpr size_t OFF_STATS = 262144;     // 5 * 128 floats (sum, sumsq)
static constexpr size_t OFF_SB    = 266240;     // 5 * 128 floats (scale, bias)
static constexpr size_t OFF_H0    = 524288;     // B*N*64 f32
static constexpr size_t OFF_H1    = 4718592;
static constexpr size_t OFF_H2    = 8912896;
static constexpr size_t OFF_E2    = 13631488;   // B*E*64 f16 (decoder raw edges)

// packed-weight offsets in halves
static constexpr size_t PK_MLP_W1 = 0;      // 128x128
static constexpr size_t PK_MLP_W2 = 16384;  // 128x64
static constexpr size_t PK_EE_W1  = 24576;  // 128x128
static constexpr size_t PK_EE_W2  = 40960;  // 128x64
static constexpr size_t PK_EN_W1  = 49152;  // 64x128
static constexpr size_t PK_EN_W2  = 57344;  // 128x64
static constexpr size_t PK_DE_W1  = 65536;  // 128x128
static constexpr size_t PK_DE_W2  = 81920;  // 128x64
static constexpr size_t PK_DN_W1  = 90112;  // 64x128
static constexpr size_t PK_DN_W2  = 98304;  // 128x64
static constexpr size_t PK_OUT_W  = 106496; // 64x64

// ---------------- fragment loads (CDNA5 wave32 WMMA layouts) ----------------
// A (16x32 f16): lane L, row M=L%16; 16 halves = K[16*(L/16)..+15]
__device__ __forceinline__ v16h ld_afrag(const _Float16* __restrict__ p) {
  h8v lo = *(const h8v*)p;
  h8v hi = *(const h8v*)(p + 8);
  v16h a;
#pragma unroll
  for (int i = 0; i < 8; ++i) { a[i] = lo[i]; a[i + 8] = hi[i]; }
  return a;
}
// B from packed panel: 32 lanes x 16 contiguous halves
__device__ __forceinline__ v16h ld_bfrag(const _Float16* __restrict__ p) {
  h8v lo = *(const h8v*)p;
  h8v hi = *(const h8v*)(p + 8);
  v16h b;
#pragma unroll
  for (int i = 0; i < 8; ++i) { b[i] = lo[i]; b[i + 8] = hi[i]; }
  return b;
}

// One GEMM layer over a 128-row LDS tile. Each of 8 waves owns a 16-row strip.
// KD = inner dim, ND = output dim; optional ELU; output to LDS (f16 or f32).
template<int KD, int ND, bool DO_ELU, bool HALF_OUT>
__device__ __forceinline__ void gemm_layer(const _Float16* __restrict__ aLds,
                                           const _Float16* __restrict__ pW,
                                           const float* __restrict__ bias,
                                           void* __restrict__ outLds) {
  const int lane = threadIdx.x & 31;
  const int wave = threadIdx.x >> 5;
  const int lrow = lane & 15;
  const int hsel = lane >> 4;
  const int m0   = wave * 16;
#pragma unroll
  for (int nb = 0; nb < ND / 16; ++nb) {
    v8f acc;
    float bv = bias[nb * 16 + lrow];
#pragma unroll
    for (int r = 0; r < 8; ++r) acc[r] = bv;
#pragma unroll
    for (int kb = 0; kb < KD / 32; ++kb) {
      v16h a = ld_afrag(aLds + (size_t)(m0 + lrow) * KD + kb * 32 + hsel * 16);
      v16h b = ld_bfrag(pW + ((size_t)(kb * (ND / 16) + nb) * 32 + lane) * 16);
      acc = __builtin_amdgcn_wmma_f32_16x16x32_f16(false, a, false, b,
                                                   (short)0, acc, false, false);
    }
#pragma unroll
    for (int r = 0; r < 8; ++r) {
      float v = acc[r];
      if (DO_ELU) v = (v > 0.f) ? v : (__expf(v) - 1.f);
      const int row = m0 + hsel * 8 + r;
      const int c   = nb * 16 + lrow;
      if (HALF_OUT) ((_Float16*)outLds)[(size_t)row * ND + c] = (_Float16)v;
      else          ((float*)outLds)[(size_t)row * ND + c]    = v;
    }
  }
}

// Store the f32 LDS out-tile (MT x 64) to global (f32 or f16) + BN stat partials.
template<bool F16OUT>
__device__ __forceinline__ void epilogue(const float* __restrict__ sOut,
                                         float* __restrict__ sRed,
                                         float* __restrict__ gF32,
                                         _Float16* __restrict__ gF16,
                                         size_t base64,
                                         float* __restrict__ gstats) {
  const int tid = threadIdx.x;
  const int c   = tid & 63;
  float s = 0.f, q = 0.f;
#pragma unroll
  for (int k = 0; k < (MT * 64) / 256; ++k) {
    const int idx = tid + 256 * k;               // == row*64 + c (256 % 64 == 0)
    const float v = sOut[idx];
    s += v; q += v * v;
    if constexpr (F16OUT) gF16[base64 + idx] = (_Float16)v;
    else                  gF32[base64 + idx] = v;
  }
  atomicAdd(&sRed[c], s);
  atomicAdd(&sRed[64 + c], q);
  __syncthreads();
  if (tid < 64) {
    atomicAdd(&gstats[tid], sRed[tid]);
    atomicAdd(&gstats[64 + tid], sRed[64 + tid]);
  }
}

// ---------------- kernels ----------------

// Pack a row-major [K,N] f32 weight matrix into f16 WMMA-B panels.
__global__ void k_pack_w(const float* __restrict__ W, _Float16* __restrict__ dst,
                         int K, int N) {
  const int i = blockIdx.x * 256 + threadIdx.x;
  if (i >= K * N) return;
  const int p    = i >> 9;          // panel (32 lanes * 16 halves)
  const int lane = (i >> 4) & 31;
  const int idx  = i & 15;
  const int nb16 = N >> 4;
  const int kb = p / nb16, nb = p - kb * nb16;
  const int k = kb * 32 + ((lane >> 4) << 4) + idx;
  const int n = nb * 16 + (lane & 15);
  dst[i] = (_Float16)W[(size_t)k * N + n];
}

__global__ void k_bn_finalize(const float* __restrict__ stats,
                              const float* __restrict__ g,
                              const float* __restrict__ beta,
                              float* __restrict__ sb, float inv_cnt) {
  const int c = threadIdx.x;
  if (c < 64) {
    const float m = stats[c] * inv_cnt;
    const float v = stats[64 + c] * inv_cnt - m * m;
    const float s = g[c] * rsqrtf(v + 1e-5f);
    sb[c] = s;
    sb[64 + c] = beta[c] - m * s;
  }
}

// Node encode: x[B*N,128] -> ELU(ELU(x W1 + b1) W2 + b2) -> h0 (raw) + stats
__global__ __launch_bounds__(256) void k_node_encode(
    const float* __restrict__ x, const _Float16* __restrict__ pW1,
    const float* __restrict__ b1, const _Float16* __restrict__ pW2,
    const float* __restrict__ b2, float* __restrict__ h0,
    float* __restrict__ gstats) {
  __shared__ _Float16 sPre[MT * 128];
  __shared__ _Float16 sHid[MT * 128];
  __shared__ float    sOut[MT * 64];
  __shared__ float    sRed[128];
  const int tid = threadIdx.x;
  if (tid < 128) sRed[tid] = 0.f;
  const size_t base = (size_t)blockIdx.x * MT * 128;
  for (int idx = tid; idx < MT * 128; idx += 256)
    sPre[idx] = (_Float16)x[base + idx];
  __syncthreads();
  gemm_layer<128, 128, true, true >(sPre, pW1, b1, sHid);
  __syncthreads();
  gemm_layer<128, 64,  true, false>(sHid, pW2, b2, sOut);
  __syncthreads();
  epilogue<false>(sOut, sRed, h0, nullptr, (size_t)blockIdx.x * MT * 64, gstats);
}

// Edge MLP (encoder OUTF16=0 writes f32 raw to d_out edge region;
//           decoder OUTF16=1 writes f16 raw to workspace).
template<int OUTF16>
__global__ __launch_bounds__(256) void k_edge(
    const float* __restrict__ hsrc, const float* __restrict__ sbIn,
    const int* __restrict__ recv, const int* __restrict__ send,
    const _Float16* __restrict__ pW1, const float* __restrict__ b1,
    const _Float16* __restrict__ pW2, const float* __restrict__ b2,
    float* __restrict__ outF32, _Float16* __restrict__ outF16,
    float* __restrict__ gstats) {
  __shared__ _Float16 sPre[MT * 128];
  __shared__ _Float16 sHid[MT * 128];
  __shared__ float    sOut[MT * 64];
  __shared__ float    sRed[128];
  const int tid = threadIdx.x;
  if (tid < 128) sRed[tid] = 0.f;
  const int R0 = blockIdx.x * MT;
  for (int idx = tid; idx < MT * 128; idx += 256) {
    const int row = idx >> 7, col = idx & 127;
    const int R = R0 + row;
    const int b = R / Ec;
    const int e = R - b * Ec;
    const int node = (col < 64) ? recv[e] : send[e];
    const int ch = col & 63;
    const float v = hsrc[((size_t)b << 12) + (node << 6) + ch];
    sPre[idx] = (_Float16)fmaf(v, sbIn[ch], sbIn[64 + ch]);
  }
  __syncthreads();
  gemm_layer<128, 128, true, true >(sPre, pW1, b1, sHid);
  __syncthreads();
  gemm_layer<128, 64,  true, false>(sHid, pW2, b2, sOut);
  __syncthreads();
  epilogue<OUTF16 != 0>(sOut, sRed, outF32, outF16,
                        (size_t)blockIdx.x * MT * 64, gstats);
}

// In-place BN apply on encoder edge activations (-> edge_weight output).
__global__ void k_bn_apply(float* __restrict__ e, const float* __restrict__ sb,
                           size_t n) {
  const size_t i = (size_t)blockIdx.x * 256 + threadIdx.x;
  if (i < n) {
    const int c = (int)(i & 63);
    e[i] = fmaf(e[i], sb[c], sb[64 + c]);
  }
}

// edge2node aggregate (+ decoder gating) fused with node MLP.
template<bool DEC>
__global__ __launch_bounds__(256) void k_node_gnn(
    const float* __restrict__ ew, const _Float16* __restrict__ raw2,
    const float* __restrict__ sb3, const _Float16* __restrict__ pW1,
    const float* __restrict__ b1, const _Float16* __restrict__ pW2,
    const float* __restrict__ b2, float* __restrict__ hout,
    float* __restrict__ gstats) {
  __shared__ _Float16 sPre[MT * 64];
  __shared__ _Float16 sHid[MT * 128];
  __shared__ float    sOut[MT * 64];
  __shared__ float    sRed[128];
  const int tid = threadIdx.x;
  if (tid < 128) sRed[tid] = 0.f;
  const int R0 = blockIdx.x * MT;
  for (int idx = tid; idx < MT * 64; idx += 256) {
    const int row = idx >> 6, col = idx & 63;
    const int R = R0 + row;
    const int b = R >> 6, n = R & 63;
    const size_t ebase = (((size_t)b * Ec) + (size_t)n * 63) * 64 + col;
    float s = 0.f;
#pragma unroll 1
    for (int k = 0; k < 63; ++k) {
      const float w = ew[ebase + (size_t)k * 64];
      if constexpr (DEC) {
        const float r2 = (float)raw2[ebase + (size_t)k * 64];
        s += fmaf(r2, sb3[col], sb3[64 + col]) * w;
      } else {
        s += w;
      }
    }
    sPre[idx] = (_Float16)(s * 0.015625f);  // /N
  }
  __syncthreads();
  gemm_layer<64,  128, true, true >(sPre, pW1, b1, sHid);
  __syncthreads();
  gemm_layer<128, 64,  true, false>(sHid, pW2, b2, sOut);
  __syncthreads();
  epilogue<false>(sOut, sRed, hout, nullptr, (size_t)blockIdx.x * MT * 64,
                  gstats);
}

// Output projection: out = BN4(h2) @ Wd + bd
__global__ __launch_bounds__(256) void k_out(
    const float* __restrict__ h2, const float* __restrict__ sb4,
    const _Float16* __restrict__ pW, const float* __restrict__ bd,
    float* __restrict__ out) {
  __shared__ _Float16 sPre[MT * 64];
  __shared__ float    sOut[MT * 64];
  const int tid = threadIdx.x;
  const size_t base = (size_t)blockIdx.x * MT * 64;
  for (int idx = tid; idx < MT * 64; idx += 256) {
    const int c = idx & 63;
    sPre[idx] = (_Float16)fmaf(h2[base + idx], sb4[c], sb4[64 + c]);
  }
  __syncthreads();
  gemm_layer<64, 64, false, false>(sPre, pW, bd, sOut);
  __syncthreads();
  for (int idx = tid; idx < MT * 64; idx += 256) out[base + idx] = sOut[idx];
}

// ---------------- host launch ----------------
extern "C" void kernel_launch(void* const* d_in, const int* in_sizes, int n_in,
                              void* d_out, int out_size, void* d_ws,
                              size_t ws_size, hipStream_t stream) {
  const float* x    = (const float*)d_in[0];
  const int*   recv = (const int*)d_in[1];
  const int*   send = (const int*)d_in[2];
  // p_mlp, p_enc_edge, p_enc_node, p_dec_edge, p_dec_node: (W1,b1,W2,b2,g,beta)
  const float* mW1 = (const float*)d_in[3];  const float* mb1 = (const float*)d_in[4];
  const float* mW2 = (const float*)d_in[5];  const float* mb2 = (const float*)d_in[6];
  const float* mg  = (const float*)d_in[7];  const float* mbe = (const float*)d_in[8];
  const float* eW1 = (const float*)d_in[9];  const float* eb1 = (const float*)d_in[10];
  const float* eW2 = (const float*)d_in[11]; const float* eb2 = (const float*)d_in[12];
  const float* eg  = (const float*)d_in[13]; const float* ebe = (const float*)d_in[14];
  const float* nW1 = (const float*)d_in[15]; const float* nb1 = (const float*)d_in[16];
  const float* nW2 = (const float*)d_in[17]; const float* nb2 = (const float*)d_in[18];
  const float* ng  = (const float*)d_in[19]; const float* nbe = (const float*)d_in[20];
  const float* dW1 = (const float*)d_in[21]; const float* db1 = (const float*)d_in[22];
  const float* dW2 = (const float*)d_in[23]; const float* db2 = (const float*)d_in[24];
  const float* dg  = (const float*)d_in[25]; const float* dbe = (const float*)d_in[26];
  const float* oW1 = (const float*)d_in[27]; const float* ob1 = (const float*)d_in[28];
  const float* oW2 = (const float*)d_in[29]; const float* ob2 = (const float*)d_in[30];
  const float* og  = (const float*)d_in[31]; const float* obe = (const float*)d_in[32];
  const float* Wd  = (const float*)d_in[33]; const float* bd  = (const float*)d_in[34];

  char* ws = (char*)d_ws;
  _Float16* pkw   = (_Float16*)(ws + OFF_PKW);
  float*    stats = (float*)(ws + OFF_STATS);
  float*    sb    = (float*)(ws + OFF_SB);
  float*    h0    = (float*)(ws + OFF_H0);
  float*    h1    = (float*)(ws + OFF_H1);
  float*    h2    = (float*)(ws + OFF_H2);
  _Float16* e2    = (_Float16*)(ws + OFF_E2);

  float* out_node = (float*)d_out;
  float* out_edge = out_node + NODE_OUT_ELEMS;

  hipMemsetAsync(stats, 0, 5 * 128 * sizeof(float), stream);

  // pack all weights into WMMA-B f16 panels
  struct PK { const float* w; int K, N; size_t off; };
  const PK pks[11] = {
      {mW1, 128, 128, PK_MLP_W1}, {mW2, 128, 64, PK_MLP_W2},
      {eW1, 128, 128, PK_EE_W1},  {eW2, 128, 64, PK_EE_W2},
      {nW1, 64, 128, PK_EN_W1},   {nW2, 128, 64, PK_EN_W2},
      {dW1, 128, 128, PK_DE_W1},  {dW2, 128, 64, PK_DE_W2},
      {oW1, 64, 128, PK_DN_W1},   {oW2, 128, 64, PK_DN_W2},
      {Wd, 64, 64, PK_OUT_W}};
  for (int i = 0; i < 11; ++i)
    k_pack_w<<<(pks[i].K * pks[i].N + 255) / 256, 256, 0, stream>>>(
        pks[i].w, pkw + pks[i].off, pks[i].K, pks[i].N);

  const float invBN = 1.f / 16384.f;        // B*N rows
  const float invBE = 1.f / 1032192.f;      // B*E rows

  // stage 0: node encode
  k_node_encode<<<128, 256, 0, stream>>>(x, pkw + PK_MLP_W1, mb1,
                                         pkw + PK_MLP_W2, mb2, h0, stats + 0);
  k_bn_finalize<<<1, 64, 0, stream>>>(stats + 0, mg, mbe, sb + 0, invBN);

  // stage 1: encoder edge MLP (raw f32 -> d_out edge region)
  k_edge<0><<<8064, 256, 0, stream>>>(h0, sb + 0, recv, send, pkw + PK_EE_W1,
                                      eb1, pkw + PK_EE_W2, eb2, out_edge,
                                      nullptr, stats + 128);
  k_bn_finalize<<<1, 64, 0, stream>>>(stats + 128, eg, ebe, sb + 128, invBE);
  k_bn_apply<<<(unsigned)((EDGE_OUT_ELEMS + 255) / 256), 256, 0, stream>>>(
      out_edge, sb + 128, EDGE_OUT_ELEMS);   // -> edge_weight (final output)

  // stage 2: encoder aggregate + node MLP
  k_node_gnn<false><<<128, 256, 0, stream>>>(out_edge, nullptr, nullptr,
                                             pkw + PK_EN_W1, nb1,
                                             pkw + PK_EN_W2, nb2, h1,
                                             stats + 256);
  k_bn_finalize<<<1, 64, 0, stream>>>(stats + 256, ng, nbe, sb + 256, invBN);

  // stage 3: decoder edge MLP (raw f16 -> workspace)
  k_edge<1><<<8064, 256, 0, stream>>>(h1, sb + 256, recv, send, pkw + PK_DE_W1,
                                      db1, pkw + PK_DE_W2, db2, nullptr, e2,
                                      stats + 384);
  k_bn_finalize<<<1, 64, 0, stream>>>(stats + 384, dg, dbe, sb + 384, invBE);

  // stage 4: decoder aggregate (BN3(raw2)*edge_weight) + node MLP
  k_node_gnn<true><<<128, 256, 0, stream>>>(out_edge, e2, sb + 384,
                                            pkw + PK_DN_W1, ob1,
                                            pkw + PK_DN_W2, ob2, h2,
                                            stats + 512);
  k_bn_finalize<<<1, 64, 0, stream>>>(stats + 512, og, obe, sb + 512, invBN);

  // stage 5: output projection
  k_out<<<128, 256, 0, stream>>>(h2, sb + 512, pkw + PK_OUT_W, bd, out_node);
}
